// GraphSAGELayers_34711925686455
// MI455X (gfx1250) — matrix operand order repaired and checked by reading, not compiled, not run on hardware
//
#include <hip/hip_runtime.h>
#include <hip/hip_bf16.h>

#define N_NODES 100000
#define N_EDGES 800000
#define DIM     128
#define LN_EPS  1e-5f

typedef __attribute__((ext_vector_type(16))) __bf16 v16bf;
typedef __attribute__((ext_vector_type(8)))  float  v8f;

// ---------------------------------------------------------------- utilities
__global__ __launch_bounds__(256) void zero_kernel(float4* __restrict__ p, int n4) {
    int i = blockIdx.x * 256 + threadIdx.x;
    if (i < n4) p[i] = make_float4(0.f, 0.f, 0.f, 0.f);
}

__global__ __launch_bounds__(256) void deg_kernel(const int* __restrict__ dst,
                                                  float* __restrict__ deg, int ne) {
    int e = blockIdx.x * 256 + threadIdx.x;
    if (e < ne) unsafeAtomicAdd(&deg[dst[e]], 1.0f);
}

__global__ __launch_bounds__(256) void invdeg_kernel(float* __restrict__ deg, int n) {
    int i = blockIdx.x * 256 + threadIdx.x;
    if (i < n) deg[i] = 1.0f / fmaxf(deg[i], 1.0f);
}

// ------------------------------------------------- mean-aggregation scatter
// One wave32 per edge; each lane moves a float4 slice (32 lanes * 16B = 512B row).
// agg fits in the 192MB L2, so the f32 atomics resolve in-cache.
__global__ __launch_bounds__(256) void scatter_kernel(const float* __restrict__ x,
                                                      const int* __restrict__ src,
                                                      const int* __restrict__ dst,
                                                      float* __restrict__ agg, int ne) {
    int e = blockIdx.x * 8 + (threadIdx.x >> 5);
    if (e >= ne) return;
    int lane = threadIdx.x & 31;
    int s = src[e];
    int d = dst[e];
    float4 v = ((const float4*)(x + (size_t)s * DIM))[lane];
    float* ap = agg + (size_t)d * DIM + lane * 4;
    unsafeAtomicAdd(ap + 0, v.x);
    unsafeAtomicAdd(ap + 1, v.y);
    unsafeAtomicAdd(ap + 2, v.z);
    unsafeAtomicAdd(ap + 3, v.w);
}

// ---------------------------------------------------- fused SAGE layer core
// out = relu(LN(mean @ Wl^T + bl + x @ Wr^T)) [+ x  if RES]
// Block = 256 threads = 8 wave32s; block owns a 16-node tile; wave w owns
// output columns [16w, 16w+16). Each wave issues 8 v_wmma_f32_16x16x32_bf16.
template <bool RES>
__global__ __launch_bounds__(256) void sage_gemm_ln_kernel(
    const float* __restrict__ agg, const float* __restrict__ inv_deg,
    const float* __restrict__ xroot,
    const float* __restrict__ Wl, const float* __restrict__ bl,
    const float* __restrict__ Wr,
    const float* __restrict__ gamma, const float* __restrict__ beta,
    float* __restrict__ out) {
    __shared__ float hbuf[16][DIM + 4];   // +4 pad: breaks stride-128 bank conflicts
    __shared__ float mu_s[16], rs_s[16];

    const int node0 = blockIdx.x * 16;
    const int tid   = threadIdx.x;
    const int wave  = tid >> 5;          // column tile 0..7
    const int lane  = tid & 31;
    const int half  = lane >> 4;         // 0 | 1
    const int lrow  = lane & 15;

    // Row this lane feeds into the A matrices (ISA 16-bit A layout, p.108)
    const int m_node = node0 + lrow;
    const float idg  = inv_deg[m_node];
    const float* aggrow = agg   + (size_t)m_node * DIM;
    const float* xrow   = xroot + (size_t)m_node * DIM;
    // Column this lane feeds into the B matrices (B = W^T -> contiguous W rows)
    const float* wlrow = Wl + (size_t)(wave * 16 + lrow) * DIM;
    const float* wrrow = Wr + (size_t)(wave * 16 + lrow) * DIM;

    __builtin_prefetch(wlrow, 0, 1);
    __builtin_prefetch(wrrow, 0, 1);

    v8f acc = {};
#pragma unroll
    for (int k0 = 0; k0 < DIM; k0 += 32) {
        v16bf a_mean, a_x, b_l, b_r;
        const int kA = k0 + half * 8;    // A: lanes<16 -> K {0..7,16..23}, else {8..15,24..31}
#pragma unroll
        for (int i = 0; i < 8; ++i) {
            a_mean[i]     = (__bf16)(aggrow[kA + i] * idg);
            a_mean[i + 8] = (__bf16)(aggrow[kA + 16 + i] * idg);
            a_x[i]        = (__bf16)(xrow[kA + i]);
            a_x[i + 8]    = (__bf16)(xrow[kA + 16 + i]);
        }
        const int kB = k0 + half * 8;    // B: lanes<16 -> K {0..7,16..23}, else {8..15,24..31}
#pragma unroll
        for (int i = 0; i < 16; ++i) {
            b_l[i] = (__bf16)(wlrow[kB + i + ((i >> 3) << 3)]);
            b_r[i] = (__bf16)(wrrow[kB + i + ((i >> 3) << 3)]);
        }
        acc = __builtin_amdgcn_wmma_f32_16x16x32_bf16(false, a_mean, false, b_l,
                                                      (short)0, acc, false, false);
        acc = __builtin_amdgcn_wmma_f32_16x16x32_bf16(false, a_x,    false, b_r,
                                                      (short)0, acc, false, false);
    }

    // C/D layout: VGPR r -> row (r + 8*half), col = lane&15 (within tile)
    const int col  = wave * 16 + lrow;
    const float bias = bl[col];
#pragma unroll
    for (int r = 0; r < 8; ++r)
        hbuf[r + half * 8][col] = acc[r] + bias;
    __syncthreads();

    if (tid < 16) {
        float s = 0.f, s2 = 0.f;
#pragma unroll 4
        for (int c = 0; c < DIM; ++c) {
            float v = hbuf[tid][c];
            s += v; s2 += v * v;
        }
        float mu  = s * (1.0f / DIM);
        float var = s2 * (1.0f / DIM) - mu * mu;
        mu_s[tid] = mu;
        rs_s[tid] = rsqrtf(var + LN_EPS);
    }
    __syncthreads();

#pragma unroll
    for (int idx = tid; idx < 16 * DIM; idx += 256) {
        int r = idx >> 7, c = idx & (DIM - 1);
        float v = (hbuf[r][c] - mu_s[r]) * rs_s[r] * gamma[c] + beta[c];
        v = fmaxf(v, 0.f);
        if (RES) v += xroot[(size_t)(node0 + r) * DIM + c];
        out[(size_t)(node0 + r) * DIM + c] = v;
    }
}

// ------------------------------------------------------------------ launch
extern "C" void kernel_launch(void* const* d_in, const int* in_sizes, int n_in,
                              void* d_out, int out_size, void* d_ws, size_t ws_size,
                              hipStream_t stream) {
    const float* x0   = (const float*)d_in[0];
    const int*   eidx = (const int*)d_in[1];
    const float* Wl   = (const float*)d_in[2];
    const float* bl   = (const float*)d_in[3];
    const float* Wr   = (const float*)d_in[4];
    const float* lng  = (const float*)d_in[5];
    const float* lnb  = (const float*)d_in[6];

    const int* src = eidx;
    const int* dst = eidx + N_EDGES;

    float* inv_deg = (float*)d_ws;                         // N
    float* agg     = inv_deg + N_NODES;                    // N*D
    float* xbuf2   = agg + (size_t)N_NODES * DIM;          // N*D (layer-1 output)
    float* xbuf1   = (float*)d_out;                        // layer-0 output lives in d_out

    const int ZB_DEG = (N_NODES / 4 + 255) / 256;
    const int ZB_AGG = ((int)((size_t)N_NODES * DIM / 4) + 255) / 256;
    const int EB     = (N_EDGES + 255) / 256;
    const int SB     = (N_EDGES + 7) / 8;                  // 8 edges (waves) per block
    const int GB     = N_NODES / 16;                       // 6250 node tiles

    // degrees -> inverse degrees (computed once, like the reference)
    zero_kernel<<<ZB_DEG, 256, 0, stream>>>((float4*)inv_deg, N_NODES / 4);
    deg_kernel<<<EB, 256, 0, stream>>>(dst, inv_deg, N_EDGES);
    invdeg_kernel<<<(N_NODES + 255) / 256, 256, 0, stream>>>(inv_deg, N_NODES);

    const size_t WSTRIDE = (size_t)DIM * DIM;

    // ---- layer 0: x0 -> xbuf1 (no residual)
    zero_kernel<<<ZB_AGG, 256, 0, stream>>>((float4*)agg, (int)((size_t)N_NODES * DIM / 4));
    scatter_kernel<<<SB, 256, 0, stream>>>(x0, src, dst, agg, N_EDGES);
    sage_gemm_ln_kernel<false><<<GB, 256, 0, stream>>>(
        agg, inv_deg, x0, Wl + 0 * WSTRIDE, bl + 0 * DIM, Wr + 0 * WSTRIDE,
        lng + 0 * DIM, lnb + 0 * DIM, xbuf1);

    // ---- layer 1: xbuf1 -> xbuf2 (residual)
    zero_kernel<<<ZB_AGG, 256, 0, stream>>>((float4*)agg, (int)((size_t)N_NODES * DIM / 4));
    scatter_kernel<<<SB, 256, 0, stream>>>(xbuf1, src, dst, agg, N_EDGES);
    sage_gemm_ln_kernel<true><<<GB, 256, 0, stream>>>(
        agg, inv_deg, xbuf1, Wl + 1 * WSTRIDE, bl + 1 * DIM, Wr + 1 * WSTRIDE,
        lng + 1 * DIM, lnb + 1 * DIM, xbuf2);

    // ---- layer 2: xbuf2 -> d_out (residual); xbuf1/d_out no longer read
    zero_kernel<<<ZB_AGG, 256, 0, stream>>>((float4*)agg, (int)((size_t)N_NODES * DIM / 4));
    scatter_kernel<<<SB, 256, 0, stream>>>(xbuf2, src, dst, agg, N_EDGES);
    sage_gemm_ln_kernel<true><<<GB, 256, 0, stream>>>(
        agg, inv_deg, xbuf2, Wl + 2 * WSTRIDE, bl + 2 * DIM, Wr + 2 * WSTRIDE,
        lng + 2 * DIM, lnb + 2 * DIM, (float*)d_out);
}
